// MultiScaleNeuralBilateralAffineTransform_31894427140451
// MI455X (gfx1250) — matrix operand order repaired and checked by reading, not compiled, not run on hardware
//
#include <hip/hip_runtime.h>
#include <hip/hip_bf16.h>

#define H_IMG 720
#define W_IMG 1280
#define NFEAT 8
#define ASTR  72   // sA row stride in halves: multiple of 8 (16B-aligned chunks), staggers banks

typedef __attribute__((ext_vector_type(16))) _Float16 v16h;
typedef __attribute__((ext_vector_type(8)))  _Float16 v8h;
typedef __attribute__((ext_vector_type(8)))  float    v8f;

__device__ __forceinline__ float fast_tanh(float x) {
#if __has_builtin(__builtin_amdgcn_tanhf)
    return __builtin_amdgcn_tanhf(x);
#else
    x = fminf(fmaxf(x, -10.0f), 10.0f);
    float e = __expf(2.0f * x);
    return (e - 1.0f) / (e + 1.0f);
#endif
}

// A fragment: 16x32 f16 tile, rows [rowbase, rowbase+16), K columns [colbase, colbase+32).
// Lane l: row = rowbase + (l&15); halves 0..15 = K colbase + hi*16 .. +15 (contiguous)
__device__ __forceinline__ v16h load_a_frag(const _Float16 (*sA)[ASTR], int rowbase, int colbase, int lane) {
    int r = lane & 15, hi = lane >> 4;
    const _Float16* p = sA[rowbase + r] + colbase;
    union { v16h v; v8h h[2]; } u;
    u.h[0] = *(const v8h*)(p + hi * 16);
    u.h[1] = *(const v8h*)(p + hi * 16 + 8);
    return u.v;
}

// B fragment from transposed weights sW[N][K]: lane l holds column N = ntile*16 + (l&15),
// K = kbase + hi*16 .. +15 (contiguous since stored [N][K]).
template<int KS>
__device__ __forceinline__ v16h load_b_frag(const _Float16 (*sW)[KS], int ntile, int kbase, int lane) {
    int n = lane & 15, hi = lane >> 4;
    const _Float16* p = sW[ntile * 16 + n] + kbase + hi * 16;
    union { v16h v; v8h h[2]; } u;
    u.h[0] = *(const v8h*)(p);
    u.h[1] = *(const v8h*)(p + 8);
    return u.v;
}

__global__ __launch_bounds__(256)
void msnbat_kernel(const float* __restrict__ rgb,
                   const float* __restrict__ g0,
                   const float* __restrict__ g1,
                   const float* __restrict__ g2,
                   const float* __restrict__ w1,
                   const float* __restrict__ w2,
                   const float* __restrict__ w3,
                   const int*   __restrict__ cam_idx_p,
                   float*       __restrict__ out)
{
    __shared__ __align__(32) _Float16 sW1[64][32];   // w1^T: [n][k], k padded 24->32
    __shared__ __align__(32) _Float16 sW2[64][64];   // w2^T: [n][k]
    __shared__ __align__(32) _Float16 sW3[16][64];   // w3^T: [n][k], n padded 12->16
    __shared__ __align__(32) _Float16 sA [256][ASTR];// per-pixel feats (K=32) then hidden (K=64)

    const int tid  = threadIdx.x;
    const int lane = tid & 31;
    const int wave = tid >> 5;
    const int cam  = cam_idx_p[0];

    // ---- stage weights (f32 -> f16, transposed, zero-padded) ----
    for (int idx = tid; idx < 64 * 32; idx += 256) {
        int n = idx >> 5, k = idx & 31;
        sW1[n][k] = (k < 24) ? (_Float16)w1[k * 64 + n] : (_Float16)0.0f;
    }
    for (int idx = tid; idx < 64 * 64; idx += 256) {
        int n = idx >> 6, k = idx & 63;
        sW2[n][k] = (_Float16)w2[k * 64 + n];
    }
    for (int idx = tid; idx < 16 * 64; idx += 256) {
        int n = idx >> 6, k = idx & 63;
        sW3[n][k] = (n < 12) ? (_Float16)w3[k * 12 + n] : (_Float16)0.0f;
    }

    // ---- per-pixel trilinear bilateral-grid slicing (24 features) ----
    const int pix = blockIdx.x * 256 + tid;
    const int py  = pix / W_IMG, px = pix % W_IMG;
    const float r = rgb[pix * 3 + 0], gc_ = rgb[pix * 3 + 1], b = rgb[pix * 3 + 2];
    const float gray = fminf(fmaxf(0.299f * r + 0.587f * gc_ + 0.114f * b, 0.0f), 1.0f);

    const float* grids[3] = { g0, g1, g2 };
    const int GX[3] = {16, 12, 8}, GY[3] = {16, 12, 8}, GL[3] = {8, 8, 4};

    #pragma unroll
    for (int s = 0; s < 3; ++s) {
        const int Wg = GX[s], Hg = GY[s], L = GL[s];
        const int HW = Hg * Wg, LHW = L * HW;
        const float* g = grids[s] + (size_t)cam * NFEAT * LHW;

        float fx = (float)px * (float)(Wg - 1) / (float)(W_IMG - 1);
        float fy = (float)py * (float)(Hg - 1) / (float)(H_IMG - 1);
        float fz = gray * (float)(L - 1);
        int x0 = (int)fx; int x1 = min(x0 + 1, Wg - 1); float wx = fx - (float)x0;
        int y0 = (int)fy; int y1 = min(y0 + 1, Hg - 1); float wy = fy - (float)y0;
        int z0 = (int)fz; int z1 = min(z0 + 1, L  - 1); float wz = fz - (float)z0;

        int o00 = (z0 * Hg + y0) * Wg, o01 = (z0 * Hg + y1) * Wg;
        int o10 = (z1 * Hg + y0) * Wg, o11 = (z1 * Hg + y1) * Wg;

        #pragma unroll
        for (int c = 0; c < NFEAT; ++c) {
            const float* gch = g + c * LHW;
            float a00 = gch[o00 + x0], a01 = gch[o00 + x1];
            float a10 = gch[o01 + x0], a11 = gch[o01 + x1];
            float b00 = gch[o10 + x0], b01 = gch[o10 + x1];
            float b10 = gch[o11 + x0], b11 = gch[o11 + x1];
            float c00 = a00 + wx * (a01 - a00);
            float c01 = a10 + wx * (a11 - a10);
            float c10 = b00 + wx * (b01 - b00);
            float c11 = b10 + wx * (b11 - b10);
            float c0  = c00 + wy * (c01 - c00);
            float c1  = c10 + wy * (c11 - c10);
            sA[tid][s * NFEAT + c] = (_Float16)(c0 + wz * (c1 - c0));
        }
    }
    #pragma unroll
    for (int k = 24; k < 32; ++k) sA[tid][k] = (_Float16)0.0f;

    __syncthreads();

    // Each wave owns 32 pixel-rows: [rb, rb+32) as two 16-row M-tiles.
    const int rb   = wave * 32;
    const int n16  = lane & 15;
    const int moff = (lane >> 4) * 8;

    // ---- layer 1: feats(32x32) @ w1(32x64) -> tanh -> hidden in sA[.,0..63] ----
    {
        v16h a0 = load_a_frag(sA, rb,      0, lane);
        v16h a1 = load_a_frag(sA, rb + 16, 0, lane);
        v8f  c[2][4];
        #pragma unroll
        for (int nt = 0; nt < 4; ++nt) {
            v16h bf = load_b_frag<32>(sW1, nt, 0, lane);
            v8f z = {};
            c[0][nt] = __builtin_amdgcn_wmma_f32_16x16x32_f16(false, a0, false, bf, (short)0, z, false, false);
            c[1][nt] = __builtin_amdgcn_wmma_f32_16x16x32_f16(false, a1, false, bf, (short)0, z, false, false);
        }
        #pragma unroll
        for (int mt = 0; mt < 2; ++mt)
            #pragma unroll
            for (int nt = 0; nt < 4; ++nt)
                #pragma unroll
                for (int i = 0; i < 8; ++i)
                    sA[rb + mt * 16 + moff + i][nt * 16 + n16] = (_Float16)fast_tanh(c[mt][nt][i]);
    }
    __builtin_amdgcn_wave_barrier();

    // ---- layer 2: hidden(32x64) @ w2(64x64) -> tanh -> sA[.,0..63] ----
    {
        v16h a[2][2];
        #pragma unroll
        for (int mt = 0; mt < 2; ++mt)
            #pragma unroll
            for (int kk = 0; kk < 2; ++kk)
                a[mt][kk] = load_a_frag(sA, rb + mt * 16, kk * 32, lane);
        v8f c[2][4];
        #pragma unroll
        for (int nt = 0; nt < 4; ++nt) {
            v16h b0 = load_b_frag<64>(sW2, nt, 0,  lane);
            v16h b1 = load_b_frag<64>(sW2, nt, 32, lane);
            #pragma unroll
            for (int mt = 0; mt < 2; ++mt) {
                v8f z = {};
                v8f acc = __builtin_amdgcn_wmma_f32_16x16x32_f16(false, a[mt][0], false, b0, (short)0, z,   false, false);
                c[mt][nt] = __builtin_amdgcn_wmma_f32_16x16x32_f16(false, a[mt][1], false, b1, (short)0, acc, false, false);
            }
        }
        #pragma unroll
        for (int mt = 0; mt < 2; ++mt)
            #pragma unroll
            for (int nt = 0; nt < 4; ++nt)
                #pragma unroll
                for (int i = 0; i < 8; ++i)
                    sA[rb + mt * 16 + moff + i][nt * 16 + n16] = (_Float16)fast_tanh(c[mt][nt][i]);
    }
    __builtin_amdgcn_wave_barrier();

    // ---- layer 3: hidden(32x64) @ w3(64x12 pad 16) -> global out ----
    {
        v16h a[2][2];
        #pragma unroll
        for (int mt = 0; mt < 2; ++mt)
            #pragma unroll
            for (int kk = 0; kk < 2; ++kk)
                a[mt][kk] = load_a_frag(sA, rb + mt * 16, kk * 32, lane);
        v16h b0 = load_b_frag<64>(sW3, 0, 0,  lane);
        v16h b1 = load_b_frag<64>(sW3, 0, 32, lane);
        #pragma unroll
        for (int mt = 0; mt < 2; ++mt) {
            v8f z = {};
            v8f acc = __builtin_amdgcn_wmma_f32_16x16x32_f16(false, a[mt][0], false, b0, (short)0, z,   false, false);
            acc     = __builtin_amdgcn_wmma_f32_16x16x32_f16(false, a[mt][1], false, b1, (short)0, acc, false, false);
            if (n16 < 12) {
                #pragma unroll
                for (int i = 0; i < 8; ++i) {
                    int row = rb + mt * 16 + moff + i;
                    size_t p = (size_t)blockIdx.x * 256 + row;
                    out[p * 12 + n16] = acc[i];
                }
            }
        }
    }
}

extern "C" void kernel_launch(void* const* d_in, const int* in_sizes, int n_in,
                              void* d_out, int out_size, void* d_ws, size_t ws_size,
                              hipStream_t stream) {
    const float* rgb = (const float*)d_in[0];
    const float* g0  = (const float*)d_in[1];
    const float* g1  = (const float*)d_in[2];
    const float* g2  = (const float*)d_in[3];
    const float* w1  = (const float*)d_in[4];
    const float* w2  = (const float*)d_in[5];
    const float* w3  = (const float*)d_in[6];
    const int*   cam = (const int*)d_in[7];
    float* out = (float*)d_out;

    dim3 grid((H_IMG * W_IMG) / 256);   // 3600 blocks, exact cover
    msnbat_kernel<<<grid, dim3(256), 0, stream>>>(rgb, g0, g1, g2, w1, w2, w3, cam, out);
}
